// GraphSageWithSampling_43009802502642
// MI455X (gfx1250) — matrix operand order repaired and checked by reading, not compiled, not run on hardware
//
#include <hip/hip_runtime.h>
#include <hip/hip_bf16.h>
#include <math.h>

// ---------------------------------------------------------------------------
// GraphSAGE forward for MI455X (gfx1250).
// GEMMs: bf16 WMMA (v_wmma_f32_16x16x32_bf16), B staged in LDS per workgroup
// via CDNA5 GLOBAL_LOAD_ASYNC_TO_LDS_B128 (ASYNCcnt) double-buffering.
// ---------------------------------------------------------------------------

typedef __attribute__((ext_vector_type(16))) __bf16 v16bf;
typedef __attribute__((ext_vector_type(8)))  float  v8f;

#define F_DIM 64
#define C_DIM 128
#define W_DIM 256
#define L_NUM 2
#define D_NUM 2

#if defined(__HIP_DEVICE_COMPILE__) && __has_builtin(__builtin_amdgcn_global_load_async_to_lds_b128)
#define ASYNC_LDS 1
#else
#define ASYNC_LDS 0
#endif

#if ASYNC_LDS
// Builtin expects int4-vector pointers: AS1 (global) source, AS3 (LDS) dest.
typedef int v4i __attribute__((vector_size(4 * sizeof(int))));
typedef __attribute__((address_space(1))) v4i as1_v4i;
typedef __attribute__((address_space(3))) v4i as3_v4i;

__device__ __forceinline__ void async_copy16(const void* g, void* l) {
    // AS1 and generic global addresses are numerically identical; the LDS
    // offset is the low 32 bits of the generic shared address (aperture rule).
    __builtin_amdgcn_global_load_async_to_lds_b128(
        (as1_v4i*)(uintptr_t)g,
        (as3_v4i*)(unsigned)(uintptr_t)l,
        0, 0);
}
__device__ __forceinline__ void wait_async0() {
#if __has_builtin(__builtin_amdgcn_s_wait_asynccnt)
    __builtin_amdgcn_s_wait_asynccnt(0);
#else
    asm volatile("s_wait_asynccnt 0x0" ::: "memory");
#endif
}
#endif

__device__ __forceinline__ float lrelu_f(float v) {
    return v >= 0.0f ? v : 0.1f * v;
}

// ---------------------------------------------------------------------------
// Weight packing: f32 row-major [K x Nc] -> bf16 WMMA-B fragments.
// dst[((kb*tilesN + nt)*32 + lane)*16 + j]:
//   column = nt*16 + (lane&15), K = kb*32 + (lane>>4)*16 + j
// Per K-block, the 4 consecutive 16-col tiles of a 64-col strip are one
// contiguous 4 KB slab -> single block-wide copy into LDS.
// ---------------------------------------------------------------------------
__global__ void pack_weights_bf16(const float* __restrict__ Wsrc,
                                  __bf16* __restrict__ dst, int K, int Nc) {
    int tid = blockIdx.x * blockDim.x + threadIdx.x;
    int total = K * Nc;
    if (tid >= total) return;
    int j    = tid & 15;
    int lane = (tid >> 4) & 31;
    int rest = tid >> 9;                 // kb*tilesN + nt
    int tilesN = Nc >> 4;
    int nt = rest % tilesN;
    int kb = rest / tilesN;
    int k   = kb * 32 + ((lane >> 4) << 4) + j;
    int col = nt * 16 + (lane & 15);
    dst[tid] = (__bf16)Wsrc[(size_t)k * Nc + col];
}

// ---------------------------------------------------------------------------
// Core GEMM:  out[M x Nc] = maybe_lrelu(A[M x K] @ Wp + bias) (+ resid)
// A fp32 row-major, split across A0/A1 at K0 (for the concat([h,h_agg]) case).
// Block = 8 waves; block tile = 256 rows x 64 cols; wave tile = 32 x 64.
// B slab (4 KB per K-step) is shared via LDS, double-buffered; all four B
// fragment pairs are loaded up front so the 8 WMMAs issue back-to-back.
// ---------------------------------------------------------------------------
__global__ void __launch_bounds__(256)
gemm_wmma_bf16(const float* __restrict__ A0, const float* __restrict__ A1,
               int K0, int K,
               const __bf16* __restrict__ Wp, const float* __restrict__ bias,
               const float* __restrict__ resid, float* __restrict__ out,
               int M, int Nc, int act) {
    __shared__ __align__(32) __bf16 ldsB[2][4 * 32 * 16];   // 2 x 4 KB

    const int tid  = threadIdx.x;
    const int lane = tid & 31;
    const int wv   = tid >> 5;

    const int tiles64 = Nc >> 6;
    const int nt64 = blockIdx.x % tiles64;
    const int mblk = blockIdx.x / tiles64;
    const int nt16 = nt64 << 2;
    const int tilesN = Nc >> 4;
    const int Mtiles = M >> 4;

    const int mt0 = (mblk * 8 + wv) * 2;        // this wave's two 16-row tiles
    const bool active = (mt0 < Mtiles);         // Mtiles is even
    const int mtc = active ? mt0 : 0;

    // A-fragment addressing (contiguous K order, matches packed B):
    //   lanes 0-15 : M=lane,    K [0..15]
    //   lanes16-31 : M=lane-16, K [16..31]
    const int half = lane >> 4;
    const int ka0  = half << 4;
    const int ka1  = (half << 4) + 8;
    const int r0 = (mtc << 4) + (lane & 15);
    const int r1 = ((mtc + 1) << 4) + (lane & 15);

    const int nkb  = K >> 5;
    const int lda0 = K0;
    const int lda1 = K - K0;

    v8f acc[8] = {};

#if ASYNC_LDS
    {   // issue stage(0): each thread copies 16 B of the 4 KB slab
        const __bf16* src = Wp + (((size_t)0 * tilesN + nt16) << 9) + tid * 8;
        async_copy16(src, &ldsB[0][tid * 8]);
    }
#endif

    for (int kb = 0; kb < nkb; ++kb) {
        const int kw = kb << 5;
        const float* Ab; int kloc, lda;
        if (kw < K0) { Ab = A0; kloc = kw;      lda = lda0; }
        else         { Ab = A1; kloc = kw - K0; lda = lda1; }
        const float* Ar0 = Ab + (size_t)r0 * lda + kloc;
        const float* Ar1 = Ab + (size_t)r1 * lda + kloc;

#if ASYNC_LDS
        wait_async0();            // my stage(kb) requests done
        __syncthreads();          // everyone's stage(kb) visible; prior reads done
        if (kb + 1 < nkb) {       // stream next slab while computing this one
            const __bf16* src =
                Wp + (((size_t)(kb + 1) * tilesN + nt16) << 9) + tid * 8;
            async_copy16(src, &ldsB[(kb + 1) & 1][tid * 8]);
        }
#else
        {
            const uint4 r =
                *(const uint4*)(Wp + (((size_t)kb * tilesN + nt16) << 9) + tid * 8);
            __syncthreads();      // prior-iteration LDS reads done
            *(uint4*)&ldsB[kb & 1][tid * 8] = r;
            __syncthreads();      // slab visible
        }
#endif

        v16bf a0, a1;
#pragma unroll
        for (int j = 0; j < 8; ++j) {
            a0[j]     = (__bf16)Ar0[ka0 + j];
            a0[8 + j] = (__bf16)Ar0[ka1 + j];
            a1[j]     = (__bf16)Ar1[ka0 + j];
            a1[8 + j] = (__bf16)Ar1[ka1 + j];
        }

        // Hoist ALL B fragments first, then issue the 8 WMMAs back-to-back
        // (single s_wait_dscnt, matrix pipe stays busy; §7.5 guidance).
        const __bf16* bb = &ldsB[kb & 1][0];
        v16bf b[4];
#pragma unroll
        for (int t = 0; t < 4; ++t)
            b[t] = *(const v16bf*)(bb + (((t << 5) + lane) << 4));
#pragma unroll
        for (int t = 0; t < 4; ++t) {
            acc[t]     = __builtin_amdgcn_wmma_f32_16x16x32_bf16(
                false, a0, false, b[t], (short)0, acc[t],     false, false);
            acc[4 + t] = __builtin_amdgcn_wmma_f32_16x16x32_bf16(
                false, a1, false, b[t], (short)0, acc[4 + t], false, false);
        }
    }

    if (!active) return;          // no barriers past this point

    // C/D layout: VGPR r -> row = mt*16 + r + 8*(lane>>4), col = nt*16+(lane&15)
    const int rb0 = (mtc << 4) + (half << 3);
    const int rb1 = ((mtc + 1) << 4) + (half << 3);
#pragma unroll
    for (int t = 0; t < 4; ++t) {
        const int col = ((nt16 + t) << 4) + (lane & 15);
        const float bs = bias ? bias[col] : 0.0f;
#pragma unroll
        for (int r = 0; r < 8; ++r) {
            float v0 = acc[t][r] + bs;
            float v1 = acc[4 + t][r] + bs;
            if (act) { v0 = lrelu_f(v0); v1 = lrelu_f(v1); }
            const size_t o0 = (size_t)(rb0 + r) * Nc + col;
            const size_t o1 = (size_t)(rb1 + r) * Nc + col;
            if (resid) { v0 += resid[o0]; v1 += resid[o1]; }
            out[o0] = v0;
            out[o1] = v1;
        }
    }
}

// ---------------------------------------------------------------------------
// Irregular graph ops
// ---------------------------------------------------------------------------
__global__ void degree_kernel(const int* __restrict__ edst,
                              float* __restrict__ deg, int E) {
    int e = blockIdx.x * blockDim.x + threadIdx.x;
    if (e < E) atomicAdd(&deg[edst[e]], 1.0f);
}

__global__ void denom_finalize(float* __restrict__ deg, int N) {
    int i = blockIdx.x * blockDim.x + threadIdx.x;
    if (i < N) deg[i] = fmaxf(deg[i] - 1.0f, 1.0f);
}

// One thread moves 4 floats of one edge: gather h[src], scatter-add h_agg[dst]
__global__ void scatter_add_h(const float* __restrict__ h,
                              const int* __restrict__ esrc,
                              const int* __restrict__ edst,
                              float* __restrict__ hagg, int E) {
    int idx = blockIdx.x * blockDim.x + threadIdx.x;
    int e = idx >> 4;
    if (e >= E) return;
    int q = (idx & 15) << 2;
    int s = esrc[e], d = edst[e];
    const float4 v = *(const float4*)(h + (size_t)s * F_DIM + q);
    float* p = hagg + (size_t)d * F_DIM + q;
    atomicAdd(p + 0, v.x);
    atomicAdd(p + 1, v.y);
    atomicAdd(p + 2, v.z);
    atomicAdd(p + 3, v.w);
}

__global__ void agg_finalize(float* __restrict__ hagg,
                             const float* __restrict__ h,
                             const float* __restrict__ denom, int N) {
    int idx = blockIdx.x * blockDim.x + threadIdx.x;
    if (idx >= N * F_DIM) return;
    int row = idx >> 6;
    hagg[idx] = (hagg[idx] - h[idx]) / denom[row];
}

__global__ void embed_add_kernel(const float* __restrict__ emb,
                                 const int* __restrict__ nid,
                                 const float* __restrict__ c,
                                 float* __restrict__ h, int N) {
    int idx = blockIdx.x * blockDim.x + threadIdx.x;
    if (idx >= N * F_DIM) return;
    int row = idx >> 6, col = idx & 63;
    h[idx] = emb[((size_t)(nid[row] + 1)) * F_DIM + col] + c[idx];
}

// One wave normalizes one 64-wide row (lane handles 2 elements).
__global__ void rownorm_kernel(const float* __restrict__ hin,
                               float* __restrict__ hout, int N) {
    int row = blockIdx.x * (blockDim.x >> 5) + (threadIdx.x >> 5);
    if (row >= N) return;
    int lane = threadIdx.x & 31;
    float a = hin[(size_t)row * F_DIM + lane];
    float b = hin[(size_t)row * F_DIM + 32 + lane];
    float s = a * a + b * b;
#pragma unroll
    for (int m = 16; m >= 1; m >>= 1) s += __shfl_xor(s, m, 32);
    float d = fmaxf(sqrtf(s), 1e-6f);
    hout[(size_t)row * F_DIM + lane]      = a / d;
    hout[(size_t)row * F_DIM + 32 + lane] = b / d;
}

// ---------------------------------------------------------------------------
// Orchestration
// ---------------------------------------------------------------------------
extern "C" void kernel_launch(void* const* d_in, const int* in_sizes, int n_in,
                              void* d_out, int out_size, void* d_ws, size_t ws_size,
                              hipStream_t stream) {
    const float* content  = (const float*)d_in[0];
    const int*   nid      = (const int*)  d_in[1];
    const int*   esrc     = (const int*)  d_in[2];
    const int*   edst     = (const int*)  d_in[3];
    const float* node_emb = (const float*)d_in[4];
    const float* proj_W   = (const float*)d_in[5];
    const float* proj_b   = (const float*)d_in[6];
    const float* proj_rW1 = (const float*)d_in[7];
    const float* proj_rb1 = (const float*)d_in[8];
    const float* proj_rW2 = (const float*)d_in[9];
    const float* proj_rb2 = (const float*)d_in[10];
    const float* Wagg     = (const float*)d_in[11];
    const float* bagg     = (const float*)d_in[12];
    const float* cW1      = (const float*)d_in[13];
    const float* cb1      = (const float*)d_in[14];
    const float* crW1     = (const float*)d_in[15];
    const float* crb1     = (const float*)d_in[16];
    const float* crW2     = (const float*)d_in[17];
    const float* crb2     = (const float*)d_in[18];
    const float* cWout    = (const float*)d_in[19];
    const float* cbout    = (const float*)d_in[20];

    const int N = in_sizes[1];
    const int E = in_sizes[2];

    // ---- workspace layout ----
    char* ws = (char*)d_ws;
    size_t o = 0;
    auto take = [&](size_t bytes) { char* p = ws + o; o += (bytes + 255) & ~(size_t)255; return p; };
    float* hbuf  = (float*)take((size_t)N * F_DIM * 4);
    float* hagg  = (float*)take((size_t)N * F_DIM * 4);
    float* habuf = (float*)take((size_t)N * F_DIM * 4);
    float* xA    = (float*)take((size_t)N * W_DIM * 4);
    float* xB    = (float*)take((size_t)N * W_DIM * 4);
    float* denom = (float*)take((size_t)N * 4);
    __bf16* pwBase = (__bf16*)take((size_t)700000 * 2);

    // ---- pack all weights to bf16 WMMA-B layout ----
    size_t woff = 0;
    auto packW = [&](const float* src, int K, int Nc) -> const __bf16* {
        __bf16* dst = pwBase + woff;
        woff += (size_t)K * Nc;
        int total = K * Nc;
        pack_weights_bf16<<<(total + 255) / 256, 256, 0, stream>>>(src, dst, K, Nc);
        return dst;
    };
    const __bf16* P_proj = packW(proj_W, C_DIM, F_DIM);
    const __bf16 *P_prW1[2], *P_prW2[2];
    for (int i = 0; i < 2; ++i) {
        P_prW1[i] = packW(proj_rW1 + (size_t)i * F_DIM * F_DIM, F_DIM, F_DIM);
        P_prW2[i] = packW(proj_rW2 + (size_t)i * F_DIM * F_DIM, F_DIM, F_DIM);
    }
    const __bf16 *P_Wagg[L_NUM], *P_cW1[L_NUM], *P_cWout[L_NUM];
    const __bf16 *P_crW1[L_NUM][D_NUM], *P_crW2[L_NUM][D_NUM];
    for (int l = 0; l < L_NUM; ++l) {
        P_Wagg[l] = packW(Wagg + (size_t)l * F_DIM * F_DIM, F_DIM, F_DIM);
        P_cW1[l]  = packW(cW1 + (size_t)l * 2 * F_DIM * W_DIM, 2 * F_DIM, W_DIM);
        for (int d = 0; d < D_NUM; ++d) {
            P_crW1[l][d] = packW(crW1 + ((size_t)l * D_NUM + d) * W_DIM * W_DIM, W_DIM, W_DIM);
            P_crW2[l][d] = packW(crW2 + ((size_t)l * D_NUM + d) * W_DIM * W_DIM, W_DIM, W_DIM);
        }
        P_cWout[l] = packW(cWout + (size_t)l * W_DIM * F_DIM, W_DIM, F_DIM);
    }

    auto gemm = [&](const float* A0, const float* A1, int K0, int K,
                    const __bf16* Wp, const float* bias, const float* resid,
                    float* out, int M, int Nc, int act) {
        int blocksM = ((M >> 4) + 15) / 16;    // 16 row-tiles per block
        int blocks  = blocksM * (Nc >> 6);
        gemm_wmma_bf16<<<blocks, 256, 0, stream>>>(A0, A1, K0, K, Wp, bias,
                                                   resid, out, M, Nc, act);
    };

    // ---- content mixer: c = lrelu(content@projW+b); 2x LinearResnet(F) ----
    float* cA = habuf;   // reuse before conv needs them
    float* cB = hagg;
    gemm(content, nullptr, C_DIM, C_DIM, P_proj, proj_b, nullptr, cA, N, F_DIM, 1);
    for (int i = 0; i < 2; ++i) {
        gemm(cA, nullptr, F_DIM, F_DIM, P_prW1[i], proj_rb1 + (size_t)i * F_DIM,
             nullptr, cB, N, F_DIM, 1);
        gemm(cB, nullptr, F_DIM, F_DIM, P_prW2[i], proj_rb2 + (size_t)i * F_DIM,
             cA, cA, N, F_DIM, 1);
    }
    // h = node_emb[nid+1] + c
    embed_add_kernel<<<(N * F_DIM + 255) / 256, 256, 0, stream>>>(node_emb, nid,
                                                                  cA, hbuf, N);

    // ---- degrees / denom ----
    (void)hipMemsetAsync(denom, 0, (size_t)N * 4, stream);
    degree_kernel<<<(E + 255) / 256, 256, 0, stream>>>(edst, denom, E);
    denom_finalize<<<(N + 255) / 256, 256, 0, stream>>>(denom, N);

    // ---- conv layers ----
    for (int l = 0; l < L_NUM; ++l) {
        const int act = (l < L_NUM - 1) ? 1 : 0;

        (void)hipMemsetAsync(hagg, 0, (size_t)N * F_DIM * 4, stream);
        scatter_add_h<<<((size_t)E * 16 + 255) / 256, 256, 0, stream>>>(
            hbuf, esrc, edst, hagg, E);
        agg_finalize<<<(N * F_DIM + 255) / 256, 256, 0, stream>>>(hagg, hbuf,
                                                                  denom, N);
        // ha = act(h_agg @ Wagg + bagg)
        gemm(hagg, nullptr, F_DIM, F_DIM, P_Wagg[l], bagg + (size_t)l * F_DIM,
             nullptr, habuf, N, F_DIM, act);
        // x = lrelu(concat([h, h_agg]) @ cW1 + cb1)
        gemm(hbuf, hagg, F_DIM, 2 * F_DIM, P_cW1[l], cb1 + (size_t)l * W_DIM,
             nullptr, xA, N, W_DIM, 1);
        // D LinearResnets at width 256
        for (int d = 0; d < D_NUM; ++d) {
            gemm(xA, nullptr, W_DIM, W_DIM, P_crW1[l][d],
                 crb1 + ((size_t)l * D_NUM + d) * W_DIM, nullptr, xB, N, W_DIM, 1);
            gemm(xB, nullptr, W_DIM, W_DIM, P_crW2[l][d],
                 crb2 + ((size_t)l * D_NUM + d) * W_DIM, xA, xA, N, W_DIM, 1);
        }
        // h_pre = ha + act(x @ cWout + cbout)
        gemm(xA, nullptr, W_DIM, W_DIM, P_cWout[l], cbout + (size_t)l * F_DIM,
             habuf, hbuf, N, F_DIM, act);
        // row L2-normalize; final layer writes straight to d_out
        float* dst = (l == L_NUM - 1) ? (float*)d_out : hbuf;
        rownorm_kernel<<<(N + 7) / 8, 256, 0, stream>>>(hbuf, dst, N);
    }
}
